// VQVAE_8486855377222
// MI455X (gfx1250) — matrix operand order repaired and checked by reading, not compiled, not run on hardware
//
#include <hip/hip_runtime.h>
#include <hip/hip_bf16.h>
#include <math.h>

// ---------------------------------------------------------------------------
// VQ-VAE forward for MI455X (gfx1250, wave32, WMMA).
// Hot spot: 131072x1024x64 distance GEMM -> v_wmma_f32_16x16x32_bf16 with
// bf16 hi/lo split (3 products) for fp32-grade accuracy. Codebook (256 KB
// bf16 hi+lo) + ||e||^2 staged into LDS with GLOBAL_LOAD_ASYNC_TO_LDS_B128
// (ASYNCcnt-tracked CDNA5 async copy), resident per workgroup (320 KB WGP LDS).
// ---------------------------------------------------------------------------

typedef __bf16 bf16_t;
typedef __attribute__((ext_vector_type(16))) __bf16 v16bf;
typedef __attribute__((ext_vector_type(8)))  __bf16 v8bf;
typedef __attribute__((ext_vector_type(8)))  float  v8f;

#define EPS_ 1e-5f
#define B_   8
#define CIN_ 3
#define H_   256
#define W_   256
#define D_   64
#define K_   1024
#define HH_  128
#define WW_  128
#define NROWS_ (B_*HH_*WW_)        // 131072
#define RGROUPS_ (NROWS_/16)       // 8192

// workspace offsets (bytes), 256-aligned
#define OFF_HRAW   0ull            // 134,217,728  (conv1 output, fp32)
#define OFF_Q      0ull            // reuse of HRAW after it is dead
#define OFF_YRAW   33554432ull     // inside HRAW region
#define OFF_FLATH  134217728ull    // 16,777,216 (flat bf16 hi)
#define OFF_FLATL  150994944ull    // 16,777,216 (flat bf16 lo)
#define OFF_FNORM  167772160ull    // 524,288
#define OFF_EH     168296448ull    // 131,072 (embed bf16 hi)
#define OFF_EL     168427520ull    // 131,072 (embed bf16 lo, contiguous w/ EH)
#define OFF_ENORM  168558592ull    // 4,096
#define OFF_IDX    168562688ull    // 524,288 (int32 indices)
#define OFF_STATS  169086976ull    // 2,048

#define VQ_LDS_BYTES 266240        // 256 KB codebook + 4 KB enorm (+ pad)

// ---------------------------------------------------------------------------
__global__ void k_init(float* stats, float* commit) {
  if (threadIdx.x < 512) stats[threadIdx.x] = 0.f;
  if (threadIdx.x == 0) *commit = 0.f;
}

__global__ void k_embed_cvt(const float* __restrict__ embed,
                            bf16_t* __restrict__ eh, bf16_t* __restrict__ el) {
  int i = blockIdx.x * 256 + threadIdx.x;
  if (i < K_ * D_) {
    float e = embed[i];
    bf16_t h = (bf16_t)e;
    eh[i] = h;
    el[i] = (bf16_t)(e - (float)h);
  }
}

__global__ void k_enorm(const float* __restrict__ embed, float* __restrict__ enorm) {
  int n = blockIdx.x * 256 + threadIdx.x;
  if (n < K_) {
    float s = 0.f;
    for (int j = 0; j < D_; ++j) { float e = embed[n*D_ + j]; s += e*e; }
    enorm[n] = s;
  }
}

// ---------------------------------------------------------------------------
// conv1: 3x3 SAME, 3->64 ch. One block per (b,y) row, thread = x column.
__global__ void k_conv1(const float* __restrict__ x, const float* __restrict__ w,
                        const float* __restrict__ bias, float* __restrict__ hraw) {
  __shared__ float xin[CIN_][3][W_ + 2];
  __shared__ float wsm[D_ * 27];
  int t = threadIdx.x;
  int b = blockIdx.x / H_;
  int y = blockIdx.x % H_;
  for (int l = t; l < CIN_*3*(W_+2); l += 256) {
    int c  = l / (3*(W_+2));
    int r  = l % (3*(W_+2));
    int ky = r / (W_+2);
    int xi = r % (W_+2);
    int iy = y + ky - 1, col = xi - 1;
    float v = 0.f;
    if (iy >= 0 && iy < H_ && col >= 0 && col < W_)
      v = x[((b*CIN_ + c)*H_ + iy)*W_ + col];
    xin[c][ky][xi] = v;
  }
  for (int l = t; l < D_*27; l += 256) wsm[l] = w[l];
  __syncthreads();
  float in[27];
#pragma unroll
  for (int j = 0; j < 27; ++j) {
    int c = j/9, ky = (j%9)/3, kx = j%3;
    in[j] = xin[c][ky][t + kx];
  }
  for (int d = 0; d < D_; ++d) {
    float acc = bias[d];
#pragma unroll
    for (int j = 0; j < 27; ++j) acc += in[j] * wsm[d*27 + j];
    hraw[((b*D_ + d)*H_ + y)*W_ + t] = acc;
  }
}

// ---------------------------------------------------------------------------
// Per-channel sum/sumsq over (N,H,W); layout [B, nchan, H*W].
__global__ void k_stats(const float* __restrict__ data, int nchan, int blocksPerChan,
                        float* __restrict__ sum, float* __restrict__ sumsq) {
  __shared__ float s1[256], s2[256];
  int chan = blockIdx.x / blocksPerChan;
  int part = blockIdx.x % blocksPerChan;
  const int HW = H_ * W_;
  int total = B_ * HW;
  int chunk = total / blocksPerChan;
  float a = 0.f, q = 0.f;
  for (int i = part*chunk + threadIdx.x; i < (part+1)*chunk; i += 256) {
    int bb = i / HW, r = i % HW;
    float v = data[(bb*nchan + chan)*HW + r];
    a += v; q += v*v;
  }
  s1[threadIdx.x] = a; s2[threadIdx.x] = q;
  __syncthreads();
  for (int o = 128; o > 0; o >>= 1) {
    if (threadIdx.x < o) { s1[threadIdx.x] += s1[threadIdx.x+o]; s2[threadIdx.x] += s2[threadIdx.x+o]; }
    __syncthreads();
  }
  if (threadIdx.x == 0) { atomicAdd(&sum[chan], s1[0]); atomicAdd(&sumsq[chan], s2[0]); }
}

__global__ void k_finalize(const float* sum, const float* sumsq,
                           float* mean, float* rstd, int nchan, float cnt) {
  int d = threadIdx.x;
  if (d < nchan) {
    float m = sum[d] / cnt;
    float v = sumsq[d] / cnt - m*m;
    mean[d] = m; rstd[d] = rsqrtf(v + EPS_);
  }
}

// ---------------------------------------------------------------------------
// BN + ReLU + 2x2 maxpool, transpose NCHW->(N*Hh*Wh, D), split into bf16 hi/lo.
__global__ void k_pool(const float* __restrict__ hraw, const float* __restrict__ mean,
                       const float* __restrict__ rstd, const float* __restrict__ gamma,
                       const float* __restrict__ beta,
                       bf16_t* __restrict__ fh, bf16_t* __restrict__ fl) {
  __shared__ float tile[D_][65];
  int t = threadIdx.x;
  int blk = blockIdx.x;
  int wtile = blk & 1;
  int hh = (blk >> 1) & 127;
  int b  = blk >> 8;
  int ww0 = wtile * 64;
  for (int p = 0; p < 16; ++p) {
    int l = p*256 + t;
    int wl = l & 63;
    int d  = l >> 6;
    int ww = ww0 + wl;
    const float* src = &hraw[((b*D_ + d)*H_ + 2*hh)*W_ + 2*ww];
    float sc = rstd[d] * gamma[d];
    float sh = beta[d] - mean[d] * sc;
    float v0 = src[0]*sc + sh,     v1 = src[1]*sc + sh;
    float v2 = src[W_]*sc + sh,    v3 = src[W_+1]*sc + sh;
    float mx = fmaxf(fmaxf(fmaxf(v0, v1), fmaxf(v2, v3)), 0.f);
    tile[d][wl] = mx;
  }
  __syncthreads();
  for (int p = 0; p < 16; ++p) {
    int l = p*256 + t;
    int d  = l & 63;
    int wl = l >> 6;
    int m = (b*HH_ + hh)*WW_ + ww0 + wl;
    float v = tile[d][wl];
    bf16_t h = (bf16_t)v;
    fh[m*D_ + d] = h;
    fl[m*D_ + d] = (bf16_t)(v - (float)h);
  }
}

// ||f||^2 per row: one wave per row, 2 elems per lane.
__global__ void k_fnorm(const bf16_t* __restrict__ fh, const bf16_t* __restrict__ fl,
                        float* __restrict__ fnorm) {
  int wid  = (blockIdx.x*256 + threadIdx.x) >> 5;
  int lane = threadIdx.x & 31;
  if (wid >= NROWS_) return;
  int base = wid*D_ + lane*2;
  float v0 = (float)fh[base]   + (float)fl[base];
  float v1 = (float)fh[base+1] + (float)fl[base+1];
  float s = v0*v0 + v1*v1;
  for (int o = 16; o > 0; o >>= 1) s += __shfl_xor(s, o, 32);
  if (lane == 0) fnorm[wid] = s;
}

// ---------------------------------------------------------------------------
// VQ argmin via WMMA. Each wave owns 16 rows; whole codebook lives in LDS,
// staged with CDNA5 async global->LDS (ASYNCcnt) instead of VGPR round-trips.
// dist = ||e||^2 - 2*f.e  (||f||^2 added only for the commit loss).
__global__ void k_vq(const bf16_t* __restrict__ fh, const bf16_t* __restrict__ fl,
                     const bf16_t* __restrict__ ehg,   // eh||el contiguous, 262144 B
                     const float* __restrict__ enorm, const float* __restrict__ fnorm,
                     int* __restrict__ idxOut, float* __restrict__ idxF,
                     float* __restrict__ commit) {
  extern __shared__ char smem[];
  bf16_t* eh  = (bf16_t*)smem;
  bf16_t* el  = eh + K_*D_;
  float*  enl = (float*)(smem + 262144);
  {
    // Async copy: 262144 B codebook + 4096 B enorm, 16 B per lane per op.
    // VDST VGPR carries the wave-relative LDS byte address (= low 32 bits of
    // the generic LDS pointer, per the flat->LDS aperture truncation rule).
    unsigned ldsBase = (unsigned)(size_t)smem;
    const char* gsrc = (const char*)ehg;
    for (int i = threadIdx.x; i < 16384; i += 256) {
      unsigned    l = ldsBase + (unsigned)i * 16u;
      const char* g = gsrc + (size_t)i * 16u;
      asm volatile("global_load_async_to_lds_b128 %0, %1, off"
                   :: "v"(l), "v"(g) : "memory");
    }
    {
      unsigned    l = ldsBase + 262144u + (unsigned)threadIdx.x * 16u;
      const char* g = (const char*)enorm + (size_t)threadIdx.x * 16u;
      asm volatile("global_load_async_to_lds_b128 %0, %1, off"
                   :: "v"(l), "v"(g) : "memory");
    }
    asm volatile("s_wait_asynccnt 0" ::: "memory");
  }
  __syncthreads();

  int lane = threadIdx.x & 31;
  int warp = threadIdx.x >> 5;
  int half = lane >> 4;        // K-half selector for A, M-half rows for C
  int l15  = lane & 15;
  int nwaves = gridDim.x * 8;
  const float cscale = 0.25f / ((float)NROWS_ * (float)D_);

  for (int rg = blockIdx.x*8 + warp; rg < RGROUPS_; rg += nwaves) {
    int m0 = rg * 16;
    int m  = m0 + l15;
    int kb = half * 16;
    const bf16_t* fhr = fh + m*D_;
    const bf16_t* flr = fl + m*D_;
    v8bf h00 = *(const v8bf*)(fhr + kb);
    v8bf h01 = *(const v8bf*)(fhr + kb + 8);
    v8bf h10 = *(const v8bf*)(fhr + 32 + kb);
    v8bf h11 = *(const v8bf*)(fhr + 32 + kb + 8);
    v8bf l00 = *(const v8bf*)(flr + kb);
    v8bf l01 = *(const v8bf*)(flr + kb + 8);
    v8bf l10 = *(const v8bf*)(flr + 32 + kb);
    v8bf l11 = *(const v8bf*)(flr + 32 + kb + 8);
    v16bf aH0 = __builtin_shufflevector(h00, h01, 0,1,2,3,4,5,6,7,8,9,10,11,12,13,14,15);
    v16bf aH1 = __builtin_shufflevector(h10, h11, 0,1,2,3,4,5,6,7,8,9,10,11,12,13,14,15);
    v16bf aL0 = __builtin_shufflevector(l00, l01, 0,1,2,3,4,5,6,7,8,9,10,11,12,13,14,15);
    v16bf aL1 = __builtin_shufflevector(l10, l11, 0,1,2,3,4,5,6,7,8,9,10,11,12,13,14,15);

    float best[8]; int bidx[8];
#pragma unroll
    for (int r = 0; r < 8; ++r) { best[r] = 3.4e38f; bidx[r] = 0; }

    for (int n0 = 0; n0 < K_; n0 += 16) {
      int n = n0 + l15;
      int sel = half * 16;
      const bf16_t* bp = eh + n*D_;
      const bf16_t* bq = el + n*D_;
      v16bf bH0 = *(const v16bf*)(bp + sel);
      v16bf bH1 = *(const v16bf*)(bp + 32 + sel);
      v16bf bL0 = *(const v16bf*)(bq + sel);
      v16bf bL1 = *(const v16bf*)(bq + 32 + sel);
      v8f acc = {0.f,0.f,0.f,0.f,0.f,0.f,0.f,0.f};
      acc = __builtin_amdgcn_wmma_f32_16x16x32_bf16(false, aH0, false, bH0, (short)0, acc, false, false);
      acc = __builtin_amdgcn_wmma_f32_16x16x32_bf16(false, aH1, false, bH1, (short)0, acc, false, false);
      acc = __builtin_amdgcn_wmma_f32_16x16x32_bf16(false, aH0, false, bL0, (short)0, acc, false, false);
      acc = __builtin_amdgcn_wmma_f32_16x16x32_bf16(false, aH1, false, bL1, (short)0, acc, false, false);
      acc = __builtin_amdgcn_wmma_f32_16x16x32_bf16(false, aL0, false, bH0, (short)0, acc, false, false);
      acc = __builtin_amdgcn_wmma_f32_16x16x32_bf16(false, aL1, false, bH1, (short)0, acc, false, false);
      float en = enl[n];
#pragma unroll
      for (int r = 0; r < 8; ++r) {
        float dist = en - 2.0f * acc[r];
        if (dist < best[r]) { best[r] = dist; bidx[r] = n; }
      }
    }
    // argmin across the 16 lanes that share each row (tie -> lowest index)
#pragma unroll
    for (int r = 0; r < 8; ++r) {
      for (int o = 1; o < 16; o <<= 1) {
        float ov = __shfl_xor(best[r], o, 32);
        int   oi = __shfl_xor(bidx[r], o, 32);
        if (ov < best[r] || (ov == best[r] && oi < bidx[r])) { best[r] = ov; bidx[r] = oi; }
      }
    }
    if (l15 == 0) {
      float csum = 0.f;
#pragma unroll
      for (int r = 0; r < 8; ++r) {
        int mm = m0 + half*8 + r;
        idxOut[mm] = bidx[r];
        idxF[mm]   = (float)bidx[r];
        csum += best[r] + fnorm[mm];   // = ||e - f||^2
      }
      atomicAdd(commit, csum * cscale);
    }
  }
}

// ---------------------------------------------------------------------------
// q[b,d,hh,ww] = embed[idx[b,hh,ww], d]   (straight-through forward == quant)
__global__ void k_scatter(const int* __restrict__ idx, const float* __restrict__ embed,
                          float* __restrict__ q) {
  int t = blockIdx.x*256 + threadIdx.x;
  if (t >= B_*D_*HH_*WW_) return;
  int ww = t & 127, hh = (t >> 7) & 127, d = (t >> 14) & 63, b = t >> 20;
  int m = (b*HH_ + hh)*WW_ + ww;
  q[t] = embed[idx[m]*D_ + d];
}

// conv2: 3x3 SAME, 64->3 ch, on nearest-x2 upsample of q (read q at iy>>1,ix>>1).
__global__ void k_conv2(const float* __restrict__ q, const float* __restrict__ w,
                        const float* __restrict__ bias, float* __restrict__ yraw) {
  __shared__ float qrow[2][D_][WW_];   // 64 KB: the two pooled rows a row needs
  int t = threadIdx.x;
  int b = blockIdx.x / H_;
  int y = blockIdx.x % H_;
  int hh_lo = (y >= 1) ? ((y-1) >> 1) : 0;
  int hh_hi = (y+1 <= H_-1) ? ((y+1) >> 1) : (HH_-1);
  for (int p = 0; p < 64; ++p) {
    int l = p*256 + t;            // 0..16383
    int ww = l & 127;
    int d  = (l >> 7) & 63;
    int rr = l >> 13;             // 0 or 1
    int hh = rr ? hh_hi : hh_lo;
    qrow[rr][d][ww] = q[((b*D_ + d)*HH_ + hh)*WW_ + ww];
  }
  __syncthreads();
  int x = t;
  float acc[CIN_];
  for (int o = 0; o < CIN_; ++o) acc[o] = bias[o];
  for (int d = 0; d < D_; ++d) {
#pragma unroll
    for (int ky = 0; ky < 3; ++ky) {
      int iy = y + ky - 1;
      if (iy < 0 || iy >= H_) continue;
      int rsel = (iy >> 1) - hh_lo;   // 0 or 1
#pragma unroll
      for (int kx = 0; kx < 3; ++kx) {
        int ix = x + kx - 1;
        if (ix < 0 || ix >= W_) continue;
        float v = qrow[rsel][d][ix >> 1];
#pragma unroll
        for (int o = 0; o < CIN_; ++o)
          acc[o] += v * w[((o*D_ + d)*3 + ky)*3 + kx];
      }
    }
  }
  for (int o = 0; o < CIN_; ++o)
    yraw[((b*CIN_ + o)*H_ + y)*W_ + x] = acc[o];
}

__global__ void k_tanh(const float* __restrict__ yraw, const float* mean, const float* rstd,
                       const float* gamma, const float* beta, float* __restrict__ out) {
  int t = blockIdx.x*256 + threadIdx.x;
  if (t >= B_*CIN_*H_*W_) return;
  int ch = (t >> 16) % CIN_;
  float v = yraw[t];
  out[t] = tanhf((v - mean[ch]) * rstd[ch] * gamma[ch] + beta[ch]);
}

// ---------------------------------------------------------------------------
extern "C" void kernel_launch(void* const* d_in, const int* in_sizes, int n_in,
                              void* d_out, int out_size, void* d_ws, size_t ws_size,
                              hipStream_t stream) {
  const float* x   = (const float*)d_in[0];
  const float* w1  = (const float*)d_in[1];
  const float* b1  = (const float*)d_in[2];
  const float* g1  = (const float*)d_in[3];
  const float* be1 = (const float*)d_in[4];
  const float* emb = (const float*)d_in[5];
  const float* w2  = (const float*)d_in[6];
  const float* b2  = (const float*)d_in[7];
  const float* g2  = (const float*)d_in[8];
  const float* be2 = (const float*)d_in[9];

  char* ws = (char*)d_ws;
  float*  hraw  = (float*)(ws + OFF_HRAW);
  float*  q     = (float*)(ws + OFF_Q);
  float*  yraw  = (float*)(ws + OFF_YRAW);
  bf16_t* fh    = (bf16_t*)(ws + OFF_FLATH);
  bf16_t* fl    = (bf16_t*)(ws + OFF_FLATL);
  float*  fnorm = (float*)(ws + OFF_FNORM);
  bf16_t* eh    = (bf16_t*)(ws + OFF_EH);
  bf16_t* el    = (bf16_t*)(ws + OFF_EL);
  float*  enorm = (float*)(ws + OFF_ENORM);
  int*    idxI  = (int*)(ws + OFF_IDX);
  float*  stats = (float*)(ws + OFF_STATS);
  float *sum1 = stats,       *sq1 = stats+64,  *mean1 = stats+128, *rstd1 = stats+192;
  float *sum2 = stats+256,   *sq2 = stats+260, *mean2 = stats+264, *rstd2 = stats+268;

  float* outY      = (float*)d_out;                       // [8,3,256,256]
  float* outIdx    = outY + (size_t)B_*CIN_*H_*W_;        // [131072] as float
  float* outCommit = outIdx + NROWS_;                     // scalar

  k_init<<<1, 512, 0, stream>>>(stats, outCommit);
  k_embed_cvt<<<(K_*D_ + 255)/256, 256, 0, stream>>>(emb, eh, el);
  k_enorm<<<(K_ + 255)/256, 256, 0, stream>>>(emb, enorm);

  k_conv1<<<B_*H_, 256, 0, stream>>>(x, w1, b1, hraw);
  k_stats<<<64*64, 256, 0, stream>>>(hraw, 64, 64, sum1, sq1);
  k_finalize<<<1, 64, 0, stream>>>(sum1, sq1, mean1, rstd1, 64, (float)(B_*H_*W_));
  k_pool<<<B_*HH_*2, 256, 0, stream>>>(hraw, mean1, rstd1, g1, be1, fh, fl);
  k_fnorm<<<NROWS_/8, 256, 0, stream>>>(fh, fl, fnorm);

  (void)hipFuncSetAttribute(reinterpret_cast<const void*>(k_vq),
                            hipFuncAttributeMaxDynamicSharedMemorySize, VQ_LDS_BYTES);
  k_vq<<<128, 256, VQ_LDS_BYTES, stream>>>(fh, fl, eh, enorm, fnorm, idxI, outIdx, outCommit);

  k_scatter<<<(B_*D_*HH_*WW_ + 255)/256, 256, 0, stream>>>(idxI, emb, q);
  k_conv2<<<B_*H_, 256, 0, stream>>>(q, w2, b2, yraw);
  k_stats<<<3*32, 256, 0, stream>>>(yraw, 3, 32, sum2, sq2);
  k_finalize<<<1, 64, 0, stream>>>(sum2, sq2, mean2, rstd2, 3, (float)(B_*H_*W_));
  k_tanh<<<(B_*CIN_*H_*W_ + 255)/256, 256, 0, stream>>>(yraw, mean2, rstd2, g2, be2, outY);
}
